// MambaVisionEncoder_6038724018450
// MI455X (gfx1250) — compile-verified, matrix operand order checked by reading, not run on hardware
//
#include <hip/hip_runtime.h>
#include <hip/hip_bf16.h>

// ---------------- model dims ----------------
#define HIDDEN 128
#define DEPTH  4
#define INTER  512
#define STATE  16
#define DT_RANK 8
#define KCONV  4
#define EPS    1e-5f
#define B_     2
#define T_     256
#define BT     (B_ * T_)            // 512 rows
#define INPUT_DIM (3 * 224 * 224)   // 150528

typedef __attribute__((ext_vector_type(16))) _Float16 v16h;
typedef __attribute__((ext_vector_type(8)))  float    v8f;

// pack 4x float4 (two contiguous 8-float runs) into a 16xf16 WMMA A/B fragment
__device__ __forceinline__ v16h pack_frag(float4 a0, float4 a1, float4 a2, float4 a3) {
    v16h v;
    v[0]  = (_Float16)a0.x; v[1]  = (_Float16)a0.y; v[2]  = (_Float16)a0.z; v[3]  = (_Float16)a0.w;
    v[4]  = (_Float16)a1.x; v[5]  = (_Float16)a1.y; v[6]  = (_Float16)a1.z; v[7]  = (_Float16)a1.w;
    v[8]  = (_Float16)a2.x; v[9]  = (_Float16)a2.y; v[10] = (_Float16)a2.z; v[11] = (_Float16)a2.w;
    v[12] = (_Float16)a3.x; v[13] = (_Float16)a3.y; v[14] = (_Float16)a3.z; v[15] = (_Float16)a3.w;
    return v;
}

// ====================================================================
// Big streaming GEMM (input projection), split-K, LDS-staged A tiles
// via gfx1250 async global->LDS copies, double buffered.
//   P[split][M][N]  = A[M, kslice] * W[N, kslice]^T      (N == 128 fixed)
// grid = (M/16, KSPLIT), block = 128 (4 waves; wave w owns cols w*32..w*32+31)
// ====================================================================

// Each thread issues two async 16B copies covering a 16x64-float A tile.
__device__ __forceinline__ void stage_async(const float* __restrict__ A, int K,
                                            int mtile, int k0, unsigned ldsbase, int tid) {
#pragma unroll
    for (int j = 0; j < 2; ++j) {
        const int f    = tid + j * 128;        // float4 index 0..255
        const int row  = f >> 4;               // 16 rows
        const int col4 = (f & 15) << 2;        // 16 float4s per row
        const float* g = A + (size_t)(mtile + row) * K + k0 + col4;
        const unsigned loff = ldsbase + (unsigned)(f << 4);
        asm volatile("global_load_async_to_lds_b128 %0, %1, off"
                     :: "v"(loff), "v"((unsigned long long)(uintptr_t)g)
                     : "memory");
    }
}

__global__ void __launch_bounds__(128)
gemm_big_wmma(const float* __restrict__ A, const float* __restrict__ W,
              float* __restrict__ P, int M, int N, int K, int kchunk) {
    __shared__ float At[2][16 * 64];           // double-buffered A tile (8 KB)

    const int tid   = threadIdx.x;
    const int lane  = tid & 31;
    const int wave  = tid >> 5;
    const int mtile = blockIdx.x * 16;
    const int split = blockIdx.y;
    const int kbeg  = split * kchunk;          // kchunk % 64 == 0

    const int r15   = lane & 15;
    const int khalf = (lane >> 4) * 8;

    // B (weights): wave covers cols wave*32 + {0..15, 16..31}; N==128 so always valid
    const float* wrow0 = W + (size_t)(wave * 32 + r15) * K + khalf;
    const float* wrow1 = wrow0 + (size_t)16 * K;

    const unsigned lds0 = (unsigned)(uintptr_t)(&At[0][0]);
    const unsigned lds1 = (unsigned)(uintptr_t)(&At[1][0]);

    v8f acc0 = {};
    v8f acc1 = {};

    const int nchunks = kchunk / 64;
    stage_async(A, K, mtile, kbeg, lds0, tid);

    for (int c = 0; c < nchunks; ++c) {
        const int buf = c & 1;
        const bool more = (c + 1 < nchunks);
        if (more)   // uniform branch: EXEC stays all-ones
            stage_async(A, K, mtile, kbeg + (c + 1) * 64, buf ? lds0 : lds1, tid);

        // retire the current buffer's (oldest, in-order) async pair
        if (more) asm volatile("s_wait_asynccnt 0x2" ::: "memory");
        else      asm volatile("s_wait_asynccnt 0x0" ::: "memory");
        __syncthreads();

        const float* As = &At[buf][0];
        const int kg = kbeg + c * 64;
#pragma unroll
        for (int sub = 0; sub < 64; sub += 32) {
            // A fragment from LDS (ds_load_b128 x4)
            const float* ar = As + r15 * 64 + sub + khalf;
            v16h af = pack_frag(*(const float4*)(ar),
                                *(const float4*)(ar + 4),
                                *(const float4*)(ar + 16),
                                *(const float4*)(ar + 20));
            __builtin_prefetch(wrow0 + kg + sub + 512, 0, 1);
            __builtin_prefetch(wrow1 + kg + sub + 512, 0, 1);
            v16h bf0 = pack_frag(*(const float4*)(wrow0 + kg + sub),
                                 *(const float4*)(wrow0 + kg + sub + 4),
                                 *(const float4*)(wrow0 + kg + sub + 16),
                                 *(const float4*)(wrow0 + kg + sub + 20));
            v16h bf1 = pack_frag(*(const float4*)(wrow1 + kg + sub),
                                 *(const float4*)(wrow1 + kg + sub + 4),
                                 *(const float4*)(wrow1 + kg + sub + 16),
                                 *(const float4*)(wrow1 + kg + sub + 20));
            acc0 = __builtin_amdgcn_wmma_f32_16x16x32_f16(false, af, false, bf0,
                                                          (short)0, acc0, false, false);
            acc1 = __builtin_amdgcn_wmma_f32_16x16x32_f16(false, af, false, bf1,
                                                          (short)0, acc1, false, false);
        }
        __syncthreads();   // all waves done reading buf before it is restaged
    }

    // store partials: VGPR r -> row mtile + r + 8*hi ; cols wave*32 + r15 (+16)
    const int hi8 = (lane >> 4) * 8;
    float* Pp = P + (size_t)split * M * N;
    const int c0 = wave * 32 + r15;
#pragma unroll
    for (int r = 0; r < 8; ++r) {
        const size_t rowoff = (size_t)(mtile + hi8 + r) * N;
        Pp[rowoff + c0]      = acc0[r];
        Pp[rowoff + c0 + 16] = acc1[r];
    }
}

// h[m,n] = bias[n] + sum_s P[s][m][n]
__global__ void __launch_bounds__(256)
reduce_bias_kernel(const float* __restrict__ P, const float* __restrict__ bias,
                   float* __restrict__ h, int KS) {
    int idx = blockIdx.x * 256 + threadIdx.x;   // BT*HIDDEN
    int n = idx & (HIDDEN - 1);
    float s = bias[n];
    for (int q = 0; q < KS; ++q) s += P[(size_t)q * BT * HIDDEN + idx];
    h[idx] = s;
}

// ====================================================================
// Generic WMMA GEMM for the small per-layer projections.
// C[M,N] = A[M,K] * W[N,K]^T (+bias[N]) (+resid[M,N]); ragged N OK.
// grid = (M/16, ceil(N/64)), block = 128.
// ====================================================================
__global__ void __launch_bounds__(128)
gemm_wmma_f16(const float* __restrict__ A, const float* __restrict__ W,
              const float* __restrict__ bias, const float* __restrict__ resid,
              float* __restrict__ C, int M, int N, int K) {
    const int lane  = threadIdx.x & 31;
    const int wave  = threadIdx.x >> 5;
    const int mtile = blockIdx.x * 16;
    const int ntile = blockIdx.y * 64 + wave * 16;

    const int r15   = lane & 15;
    const int khalf = (lane >> 4) * 8;

    const float* arow = A + (size_t)(mtile + r15) * K + khalf;
    const int    col  = ntile + r15;
    const bool   cok  = (col < N);
    const float* wrow = W + (size_t)(cok ? col : 0) * K + khalf;

    v8f acc = {};

    for (int k0 = 0; k0 < K; k0 += 32) {
        __builtin_prefetch(arow + k0 + 512, 0, 1);
        __builtin_prefetch(wrow + k0 + 512, 0, 1);
        v16h af = pack_frag(*(const float4*)(arow + k0),
                            *(const float4*)(arow + k0 + 4),
                            *(const float4*)(arow + k0 + 16),
                            *(const float4*)(arow + k0 + 20));
        v16h bf = pack_frag(*(const float4*)(wrow + k0),
                            *(const float4*)(wrow + k0 + 4),
                            *(const float4*)(wrow + k0 + 16),
                            *(const float4*)(wrow + k0 + 20));
        acc = __builtin_amdgcn_wmma_f32_16x16x32_f16(false, af, false, bf,
                                                     (short)0, acc, false, false);
    }

    const int hi8  = (lane >> 4) * 8;
    const int ccol = ntile + r15;
    if (ccol < N) {
        const float bv = bias ? bias[ccol] : 0.0f;
#pragma unroll
        for (int r = 0; r < 8; ++r) {
            size_t idx = (size_t)(mtile + hi8 + r) * N + ccol;
            float v = acc[r] + bv;
            if (resid) v += resid[idx];
            C[idx] = v;
        }
    }
}

// one block (128 threads) per row of 128 elements
__global__ void __launch_bounds__(128)
rmsnorm_kernel(const float* __restrict__ x, const float* __restrict__ w,
               float* __restrict__ o) {
    __shared__ float red[128];
    const int tid = threadIdx.x;
    const size_t base = (size_t)blockIdx.x * HIDDEN;
    float v = x[base + tid];
    red[tid] = v * v;
    __syncthreads();
#pragma unroll
    for (int s = 64; s > 0; s >>= 1) {
        if (tid < s) red[tid] += red[tid + s];
        __syncthreads();
    }
    float scale = rsqrtf(red[0] * (1.0f / HIDDEN) + EPS);
    o[base + tid] = v * scale * w[tid];
}

// causal depthwise conv1d (K=4) + bias + SiLU
__global__ void __launch_bounds__(256)
conv_silu_kernel(const float* __restrict__ proj, const float* __restrict__ cw,
                 const float* __restrict__ cb, float* __restrict__ u) {
    int idx = blockIdx.x * 256 + threadIdx.x;       // BT*INTER
    int i  = idx & (INTER - 1);
    int bt = idx >> 9;
    int t  = bt & (T_ - 1);
    int b  = bt >> 8;
    float acc = cb[i];
#pragma unroll
    for (int k = 0; k < KCONV; ++k) {
        int tt = t - (KCONV - 1) + k;
        if (tt >= 0)
            acc += cw[i * KCONV + k] * proj[((size_t)(b * T_ + tt)) * (2 * INTER) + i];
    }
    u[idx] = acc * (1.0f / (1.0f + __expf(-acc)));
}

// dt = softplus(ssm_in[:, :8] @ dtw^T + dtb)
__global__ void __launch_bounds__(256)
dtproj_kernel(const float* __restrict__ ssm_in, const float* __restrict__ dtw,
              const float* __restrict__ dtb, float* __restrict__ dt) {
    int idx = blockIdx.x * 256 + threadIdx.x;
    int i = idx & (INTER - 1);
    int m = idx >> 9;
    const float* r = ssm_in + (size_t)m * (DT_RANK + 2 * STATE);
    const float* w = dtw + (size_t)i * DT_RANK;
    float s = dtb[i];
#pragma unroll
    for (int k = 0; k < DT_RANK; ++k) s += r[k] * w[k];
    dt[idx] = (s > 20.0f) ? s : log1pf(__expf(s));
}

// sequential selective scan; one thread per (b,i), 16 states in regs;
// fuses +u*D and *silu(gate).
__global__ void __launch_bounds__(256)
ssm_scan_kernel(const float* __restrict__ dt, const float* __restrict__ u,
                const float* __restrict__ ssm_in, const float* __restrict__ proj,
                const float* __restrict__ A_log, const float* __restrict__ Dp,
                float* __restrict__ y) {
    const int ch = blockIdx.x * 256 + threadIdx.x;
    const int b = ch / INTER;
    const int i = ch & (INTER - 1);

    float a[STATE], s[STATE];
#pragma unroll
    for (int n = 0; n < STATE; ++n) {
        a[n] = -__expf(A_log[(size_t)i * STATE + n]);
        s[n] = 0.0f;
    }
    const float D = Dp[i];

    __shared__ float Bs[STATE], Cs[STATE];
    for (int t = 0; t < T_; ++t) {
        const int bt = b * T_ + t;
        if (threadIdx.x < STATE)
            Bs[threadIdx.x] = ssm_in[(size_t)bt * (DT_RANK + 2 * STATE) + DT_RANK + threadIdx.x];
        else if (threadIdx.x < 2 * STATE)
            Cs[threadIdx.x - STATE] =
                ssm_in[(size_t)bt * (DT_RANK + 2 * STATE) + DT_RANK + STATE + (threadIdx.x - STATE)];
        __syncthreads();

        const float dtv = dt[(size_t)bt * INTER + i];
        const float uv  = u[(size_t)bt * INTER + i];
        float yv = 0.0f;
#pragma unroll
        for (int n = 0; n < STATE; ++n) {
            s[n] = __expf(dtv * a[n]) * s[n] + dtv * Bs[n] * uv;
            yv += s[n] * Cs[n];
        }
        const float g = proj[(size_t)bt * (2 * INTER) + INTER + i];
        yv = (yv + uv * D) * (g * (1.0f / (1.0f + __expf(-g))));
        y[(size_t)bt * INTER + i] = yv;
        __syncthreads();
    }
}

extern "C" void kernel_launch(void* const* d_in, const int* in_sizes, int n_in,
                              void* d_out, int out_size, void* d_ws, size_t ws_size,
                              hipStream_t stream) {
    const float* x          = (const float*)d_in[0];
    const float* project_w  = (const float*)d_in[1];
    const float* project_b  = (const float*)d_in[2];
    const float* in_proj_w  = (const float*)d_in[3];
    const float* conv_w     = (const float*)d_in[4];
    const float* conv_b     = (const float*)d_in[5];
    const float* x_proj_w   = (const float*)d_in[6];
    const float* dt_proj_w  = (const float*)d_in[7];
    const float* dt_proj_b  = (const float*)d_in[8];
    const float* A_log      = (const float*)d_in[9];
    const float* D_ssm      = (const float*)d_in[10];
    const float* out_proj_w = (const float*)d_in[11];
    const float* norm_w     = (const float*)d_in[12];
    const float* norm_f_w   = (const float*)d_in[13];

    float* ws   = (float*)d_ws;
    float* h    = ws;                       // [BT, 128]
    float* hn   = h    + BT * HIDDEN;       // [BT, 128]
    float* proj = hn   + BT * HIDDEN;       // [BT, 1024]
    float* u    = proj + BT * 2 * INTER;    // [BT, 512]
    float* si   = u    + BT * INTER;        // [BT, 40]
    float* dt   = si   + BT * (DT_RANK + 2 * STATE);  // [BT, 512]
    float* y    = dt   + BT * INTER;        // [BT, 512]
    float* P    = y    + BT * INTER;        // [KS, BT, 128] split-K partials
    const size_t base_floats = (size_t)(P - ws);
    (void)in_sizes; (void)n_in; (void)out_size;

    // split-K factor for the 385 MB streaming GEMM (deterministic: partials + reduce)
    int KS = 8;                                       // 150528/8 = 18816 (mult of 64)
    if (ws_size < (base_floats + (size_t)KS * BT * HIDDEN) * sizeof(float)) KS = 1;
    const int kchunk = INPUT_DIM / KS;

    gemm_big_wmma<<<dim3(BT / 16, KS), 128, 0, stream>>>(
        x, project_w, P, BT, HIDDEN, INPUT_DIM, kchunk);
    reduce_bias_kernel<<<(BT * HIDDEN) / 256, 256, 0, stream>>>(P, project_b, h, KS);

    for (int l = 0; l < DEPTH; ++l) {
        rmsnorm_kernel<<<BT, 128, 0, stream>>>(h, norm_w + l * HIDDEN, hn);

        gemm_wmma_f16<<<dim3(BT / 16, (2 * INTER) / 64), 128, 0, stream>>>(
            hn, in_proj_w + (size_t)l * 2 * INTER * HIDDEN, nullptr, nullptr,
            proj, BT, 2 * INTER, HIDDEN);

        conv_silu_kernel<<<(BT * INTER) / 256, 256, 0, stream>>>(
            proj, conv_w + (size_t)l * INTER * KCONV, conv_b + (size_t)l * INTER, u);

        gemm_wmma_f16<<<dim3(BT / 16, 1), 128, 0, stream>>>(
            u, x_proj_w + (size_t)l * (DT_RANK + 2 * STATE) * INTER, nullptr, nullptr,
            si, BT, DT_RANK + 2 * STATE, INTER);

        dtproj_kernel<<<(BT * INTER) / 256, 256, 0, stream>>>(
            si, dt_proj_w + (size_t)l * INTER * DT_RANK, dt_proj_b + (size_t)l * INTER, dt);

        ssm_scan_kernel<<<(B_ * INTER) / 256, 256, 0, stream>>>(
            dt, u, si, proj, A_log + (size_t)l * INTER * STATE, D_ssm + (size_t)l * INTER, y);

        gemm_wmma_f16<<<dim3(BT / 16, HIDDEN / 64), 128, 0, stream>>>(
            y, out_proj_w + (size_t)l * HIDDEN * INTER, nullptr, h, h, BT, HIDDEN, INTER);
    }

    rmsnorm_kernel<<<BT, 128, 0, stream>>>(h, norm_f_w, (float*)d_out);
}
